// subRelationUnit_8452495638809
// MI455X (gfx1250) — matrix operand order repaired and checked by reading, not compiled, not run on hardware
//
#include <hip/hip_runtime.h>

typedef __attribute__((ext_vector_type(16))) _Float16 v16h;
typedef __attribute__((ext_vector_type(8)))  float    v8f;

#define NROWS 8192
#define DDIM  1024
#define KDIM  64

// ---------------------------------------------------------------------------
// Kernel A: fold WP into WK/WQ:  u_a[d] = sum_k WK_w[d,k]*WP_w[k], etc.
// consts = { a_bias, b_bias, g_bias } (tiny serial tail on thread 0)
// ---------------------------------------------------------------------------
__global__ void sru_prep_kernel(const float* __restrict__ WK_w,
                                const float* __restrict__ WQ_w,
                                const float* __restrict__ WP_w,
                                const float* __restrict__ WK_b,
                                const float* __restrict__ WQ_b,
                                const float* __restrict__ gfeat,
                                const float* __restrict__ WG_w,
                                const float* __restrict__ WG_b,
                                float* __restrict__ u_a,
                                float* __restrict__ u_b,
                                float* __restrict__ consts) {
  int d = blockIdx.x * blockDim.x + threadIdx.x;
  if (d < DDIM) {
    float sa = 0.f, sb = 0.f;
    for (int k = 0; k < KDIM; ++k) {
      float p = WP_w[k];
      sa += WK_w[d * KDIM + k] * p;
      sb += WQ_w[d * KDIM + k] * p;
    }
    u_a[d] = sa;
    u_b[d] = sb;
  }
  if (blockIdx.x == 0 && threadIdx.x == 0) {
    float ac = 0.f, bc = 0.f, gc = 0.f;
    for (int k = 0; k < KDIM; ++k) {
      ac += WK_b[k] * WP_w[k];
      bc += WQ_b[k] * WP_w[k];
    }
    for (int dd = 0; dd < DDIM; ++dd) gc += gfeat[dd] * WG_w[dd];
    consts[0] = ac;
    consts[1] = bc;
    consts[2] = gc + WG_b[0];
  }
}

// ---------------------------------------------------------------------------
// Kernel B: one wave per row; a[i], b[i], g[i] via lane-strided dots + shfl.
// ---------------------------------------------------------------------------
__global__ void __launch_bounds__(256)
sru_rowscalar_kernel(const float* __restrict__ pfeat,
                     const float* __restrict__ u_a,
                     const float* __restrict__ u_b,
                     const float* __restrict__ WG_w,
                     const float* __restrict__ consts,
                     float* __restrict__ Arow,
                     float* __restrict__ Brow,
                     float* __restrict__ Grow) {
  int wave = (blockIdx.x * blockDim.x + threadIdx.x) >> 5;
  int lane = threadIdx.x & 31;
  if (wave >= NROWS) return;
  const float* row = pfeat + (size_t)wave * DDIM;
  float sa = 0.f, sb = 0.f, sg = 0.f;
  for (int d = lane; d < DDIM; d += 32) {
    float x = row[d];
    sa += x * u_a[d];
    sb += x * u_b[d];
    sg += x * WG_w[d];
  }
  for (int m = 16; m >= 1; m >>= 1) {
    sa += __shfl_xor(sa, m, 32);
    sb += __shfl_xor(sb, m, 32);
    sg += __shfl_xor(sg, m, 32);
  }
  if (lane == 0) {
    Arow[wave] = sa + consts[0];
    Brow[wave] = sb + consts[1];
    float g = sg + consts[2];
    Grow[wave] = g > 0.f ? g : 0.f;
  }
}

// ---------------------------------------------------------------------------
// Kernel C: V = pfeat @ WV_w + WV_b, stored f16 in FRAGMENT layout:
//   for i-block blk=i/32, k-subtile kt=k/16:
//     il=i%32 -> (e = il%16, hi = il/16); lane = hi*16 + k%16
//     vhf[ ((blk*4 + kt)*32 + lane)*16 + e ]
// so kernel F can fetch a whole B fragment as 2x global_load_b128 per lane.
// 8 waves/block: 2 row-subtiles x 4 col-subtiles = 32 rows x 64 cols / block.
// ---------------------------------------------------------------------------
__global__ void __launch_bounds__(256)
sru_vproj_kernel(const float* __restrict__ pfeat,
                 const float* __restrict__ WV_w,
                 const float* __restrict__ WV_b,
                 _Float16* __restrict__ vhf) {
  const int t    = threadIdx.x;
  const int lane = t & 31;
  const int w    = t >> 5;     // wave 0..7
  const int rt   = w >> 2;     // row subtile 0..1
  const int ct   = w & 3;      // col subtile 0..3
  const int rbase = blockIdx.x * 32 + rt * 16;
  const int k0    = ct * 16;
  const int ln    = lane & 15;
  const int hi    = lane >> 4; // 0 or 1

  const float* arow = pfeat + (size_t)(rbase + ln) * DDIM;
  v8f acc = {};
  for (int kb = 0; kb < DDIM; kb += 32) {
    v16h af, bf;
    // A 16x32 f16 layout: K = kb + e + 8*((e>=8) + (lane>=16)), M = lane%16
#pragma unroll
    for (int e = 0; e < 16; e += 2) {
      int off = kb + e + 8 * ((e >> 3) + hi);
      float2 p = *(const float2*)(arow + off);
      af[e]     = (_Float16)p.x;
      af[e + 1] = (_Float16)p.y;
    }
    // B 32x16 f16 layout: K = kb + e + 8*((e>=8) + (lane>=16)), N = lane%16
#pragma unroll
    for (int e = 0; e < 16; ++e) {
      int kk = kb + e + 8 * ((e >> 3) + hi);
      bf[e] = (_Float16)WV_w[kk * KDIM + k0 + ln];
    }
    acc = __builtin_amdgcn_wmma_f32_16x16x32_f16(false, af, false, bf,
                                                 (short)0, acc, false, false);
  }
  float bias = WV_b[k0 + ln];
#pragma unroll
  for (int r = 0; r < 8; ++r) {
    int i = rbase + r + 8 * hi;      // C layout: M = r + 8*(lane>=16)
    int blk = i >> 5, il = i & 31;
    int ef = il & 15, hif = il >> 4;
    size_t addr = (((size_t)blk * 4 + ct) * 32 + (hif * 16 + ln)) * 16 + ef;
    vhf[addr] = (_Float16)(acc[r] + bias);
  }
}

// ---------------------------------------------------------------------------
// Kernel D: b_max over all rows (single block).
// ---------------------------------------------------------------------------
__global__ void sru_bmax_kernel(const float* __restrict__ Brow,
                                float* __restrict__ consts) {
  __shared__ float red[256];
  float m = -3.4e38f;
  for (int j = threadIdx.x; j < NROWS; j += 256) m = fmaxf(m, Brow[j]);
  red[threadIdx.x] = m;
  __syncthreads();
  for (int s = 128; s >= 1; s >>= 1) {
    if (threadIdx.x < s) red[threadIdx.x] = fmaxf(red[threadIdx.x], red[threadIdx.x + s]);
    __syncthreads();
  }
  if (threadIdx.x == 0) consts[3] = red[0];
}

// ---------------------------------------------------------------------------
// Kernel E: softmax row stats. g>=0 and relu monotone => row max at b_max:
//   M[i] = g*relu(a+bmax+c),  S[i] = sum_j exp(g*relu(a+b[j]+c) - M).
// Full b[] staged in 32KB LDS. Emits packed per-row float4 {a, g, M, 1/S}.
// ---------------------------------------------------------------------------
__global__ void __launch_bounds__(256)
sru_stats_kernel(const float* __restrict__ Arow,
                 const float* __restrict__ Brow,
                 const float* __restrict__ Grow,
                 const float* __restrict__ consts,
                 const float* __restrict__ WP_b,
                 float4* __restrict__ scal4) {
  __shared__ float bsh[NROWS];
  const float c    = WP_b[0];
  const float bmax = consts[3];
  for (int j = threadIdx.x; j < NROWS; j += 256) bsh[j] = Brow[j];
  __syncthreads();
  int i = blockIdx.x * 256 + threadIdx.x;
  float a = Arow[i], g = Grow[i];
  float m = g * fmaxf(a + bmax + c, 0.f);
  float s = 0.f;
  for (int j = 0; j < NROWS; ++j) {
    float tt = g * fmaxf(a + bsh[j] + c, 0.f);
    s += __expf(tt - m);
  }
  scal4[i] = make_float4(a, g, m, 1.f / s);
}

// ---------------------------------------------------------------------------
// Kernel F: out[j,k] = sum_i w[i,j] * v[i,k],
//   w[i,j] = Sinv[i] * exp(g[i]*relu(a[i]+b[j]+c) - M[i]).
// Per 32-i step: builders compute the 32x32 f16 weight tile directly into an
// A-fragment-ordered, double-buffered LDS array (one exp per element), so each
// wave's A fragment is 2x ds_load_b128 and its B fragment 2x global_load_b128.
// One barrier per step. 8 waves = 2 j-subtiles x 4 k-subtiles (32 j x 64 k).
// ---------------------------------------------------------------------------
__global__ void __launch_bounds__(256)
sru_attn_kernel(const float* __restrict__ Brow,
                const float4* __restrict__ scal4,
                const float* __restrict__ WP_b,
                const _Float16* __restrict__ vhf,
                float* __restrict__ out) {
  // whf[buf][jt][lane][e] : lane = hi*16 + j%16, e = i_local%16, hi = i_local/16
  __shared__ __attribute__((aligned(16))) _Float16 whf[2][2][32][24];
  __shared__ float bj[32];

  const int t    = threadIdx.x;
  const int lane = t & 31;
  const int w    = t >> 5;     // wave 0..7
  const int jt   = w >> 2;     // j subtile 0..1
  const int kt   = w & 3;      // k subtile 0..3
  const int jb   = blockIdx.x * 32;
  const int ln   = lane & 15;
  const int hi   = lane >> 4;
  const float c  = WP_b[0];

  if (t < 32) bj[t] = Brow[jb + t];

  // builder role: i_local = wi, j_locals wj..wj+3 (same jt group)
  const int wi  = t >> 3;          // 0..31
  const int wj  = (t & 7) * 4;     // 0..28
  const int eb  = wi & 15;                    // fragment element for i_local
  const int hib = wi >> 4;                    // fragment lane-half for i_local
  const int jtb = wj >> 4;                    // builder's j subtile
  const int lnb = wj & 15;                    // builder's first column in it
  __syncthreads();

  // prologue: build tile for ib = 0 into buffer 0
  {
    float4 s = scal4[wi];
#pragma unroll
    for (int q = 0; q < 4; ++q) {
      float tt = s.y * fmaxf(s.x + bj[wj + q] + c, 0.f);
      whf[0][jtb][hib * 16 + lnb + q][eb] = (_Float16)(s.w * __expf(tt - s.z));
    }
  }
  __syncthreads();

  v8f acc = {};
  int p = 0;
  for (int ib = 0; ib < NROWS; ib += 32) {
    // ---- consume buffer p: A fragment via 2x ds_load_b128 ----
    union { float4 q[2]; v16h h; } ua, ub;
    ua.q[0] = *(const float4*)&whf[p][jt][lane][0];
    ua.q[1] = *(const float4*)&whf[p][jt][lane][8];
    // ---- B fragment via 2x global_load_b128 (fragment-ordered vhf) ----
    const _Float16* vb = vhf + (((size_t)(ib >> 5) * 4 + kt) * 32 + lane) * 16;
    ub.q[0] = *(const float4*)(vb + 0);
    ub.q[1] = *(const float4*)(vb + 8);
    acc = __builtin_amdgcn_wmma_f32_16x16x32_f16(false, ua.h, false, ub.h,
                                                 (short)0, acc, false, false);
    // ---- build next tile into the other buffer ----
    int nxt = ib + 32;
    if (nxt < NROWS) {
      float4 s = scal4[nxt + wi];
#pragma unroll
      for (int q = 0; q < 4; ++q) {
        float tt = s.y * fmaxf(s.x + bj[wj + q] + c, 0.f);
        whf[p ^ 1][jtb][hib * 16 + lnb + q][eb] = (_Float16)(s.w * __expf(tt - s.z));
      }
    }
    __syncthreads();
    p ^= 1;
  }
#pragma unroll
  for (int r = 0; r < 8; ++r) {
    int jrow = jb + jt * 16 + r + 8 * hi;   // C layout: M = r + 8*(lane>=16)
    out[(size_t)jrow * KDIM + kt * 16 + ln] = acc[r];
  }
}

// ---------------------------------------------------------------------------
extern "C" void kernel_launch(void* const* d_in, const int* in_sizes, int n_in,
                              void* d_out, int out_size, void* d_ws, size_t ws_size,
                              hipStream_t stream) {
  const float* pfeat = (const float*)d_in[0];
  const float* gfeat = (const float*)d_in[1];
  const float* WG_w  = (const float*)d_in[2];
  const float* WG_b  = (const float*)d_in[3];
  const float* WK_w  = (const float*)d_in[4];
  const float* WK_b  = (const float*)d_in[5];
  const float* WQ_w  = (const float*)d_in[6];
  const float* WQ_b  = (const float*)d_in[7];
  const float* WP_w  = (const float*)d_in[8];
  const float* WP_b  = (const float*)d_in[9];
  const float* WV_w  = (const float*)d_in[10];
  const float* WV_b  = (const float*)d_in[11];
  float* out = (float*)d_out;

  float* wsf    = (float*)d_ws;
  float* u_a    = wsf;                      // 1024
  float* u_b    = wsf + 1024;               // 1024
  float* consts = wsf + 2048;               // 8 (a_bias, b_bias, g_bias, bmax)
  float* Arow   = wsf + 2056;               // 8192
  float* Brow   = Arow + NROWS;             // 8192
  float* Grow   = Brow + NROWS;             // 8192
  float4* scal4 = (float4*)(Grow + NROWS);  // 8192 x float4 (16B aligned)
  _Float16* vhf = (_Float16*)((float*)scal4 + 4 * NROWS);  // 8192*64 halves, 32B aligned

  sru_prep_kernel<<<4, 256, 0, stream>>>(WK_w, WQ_w, WP_w, WK_b, WQ_b,
                                         gfeat, WG_w, WG_b, u_a, u_b, consts);
  sru_rowscalar_kernel<<<NROWS / 8, 256, 0, stream>>>(pfeat, u_a, u_b, WG_w,
                                                      consts, Arow, Brow, Grow);
  sru_vproj_kernel<<<NROWS / 32, 256, 0, stream>>>(pfeat, WV_w, WV_b, vhf);
  sru_bmax_kernel<<<1, 256, 0, stream>>>(Brow, consts);
  sru_stats_kernel<<<NROWS / 256, 256, 0, stream>>>(Arow, Brow, Grow, consts,
                                                    WP_b, scal4);
  sru_attn_kernel<<<NROWS / 32, 256, 0, stream>>>(Brow, scal4, WP_b, vhf, out);
}
